// MambaIntegerBlockV2Chunked_32959579030376
// MI455X (gfx1250) — matrix operand order repaired and checked by reading, not compiled, not run on hardware
//
#include <hip/hip_runtime.h>
#include <hip/hip_bf16.h>
#include <math.h>

#define DMODEL 768
#define NHEADS 24
#define DHEAD  64
#define DSTATE 64
#define CHUNK  64
#define DINNER 1536
#define BSZ    2
#define SEQL   8192
#define NCHUNK (SEQL/CHUNK)          // 128
#define NTOK   (BSZ*SEQL)            // 16384
#define PROJ   (NHEADS*(1+2*DSTATE)) // 3096
#define NTILES (BSZ*NHEADS*NCHUNK)   // 6144
#define LDT    72                    // padded LDS tile stride (halves), 16B-aligned rows

typedef __attribute__((ext_vector_type(16))) _Float16 v16h;
typedef __attribute__((ext_vector_type(8)))  _Float16 v8h;
typedef __attribute__((ext_vector_type(8)))  float    v8f;

// ---------------- WMMA fragment helpers (16x16x32 f16, wave32) ----------------
// A-operand fragment from row-major [row][k] storage: two contiguous 8-half runs.
__device__ __forceinline__ v16h frag_row(const _Float16* s, int ld, int row0, int k0, int lane) {
  int r = row0 + (lane & 15);
  int off = (lane >> 4) << 3;
  const _Float16* p = s + r * ld + k0 + off;
  v8h lo = *(const v8h*)p;
  v8h hi = *(const v8h*)(p + 16);
  return __builtin_shufflevector(lo, hi, 0, 1, 2, 3, 4, 5, 6, 7, 8, 9, 10, 11, 12, 13, 14, 15);
}
// B-operand fragment from [n][k] storage (k contiguous): two contiguous 8-half runs.
__device__ __forceinline__ v16h frag_kc(const _Float16* s, int ld, int k0, int n0, int lane) {
  int n = n0 + (lane & 15);
  int kb = k0 + ((lane >> 4) << 3);
  const _Float16* p = s + n * ld + kb;
  v8h lo = *(const v8h*)p;
  v8h hi = *(const v8h*)(p + 16);
  return __builtin_shufflevector(lo, hi, 0, 1, 2, 3, 4, 5, 6, 7, 8, 9, 10, 11, 12, 13, 14, 15);
}

__device__ __forceinline__ v8f wmma_f32(v16h a, v16h b, v8f c) {
  return __builtin_amdgcn_wmma_f32_16x16x32_f16(false, a, false, b, (short)0, c, false, false);
}

__device__ __forceinline__ void copy8(const _Float16* g, _Float16* s) { *(v8h*)s = *(const v8h*)g; }

// ---------------- small utility kernels ----------------
__global__ void k_f32_to_f16(const float* __restrict__ src, _Float16* __restrict__ dst, int n) {
  int i = blockIdx.x * blockDim.x + threadIdx.x;
  if (i < n) dst[i] = (_Float16)src[i];
}

__global__ void __launch_bounds__(256) k_rmsnorm(const float* __restrict__ x,
                                                 const float* __restrict__ w,
                                                 _Float16* __restrict__ out) {
  int row = blockIdx.x;
  const float* xr = x + (size_t)row * DMODEL;
  __shared__ float red[8];
  float ss = 0.f;
  for (int i = threadIdx.x; i < DMODEL; i += 256) { float v = xr[i]; ss += v * v; }
  for (int m = 16; m > 0; m >>= 1) ss += __shfl_xor(ss, m, 32);
  int wid = threadIdx.x >> 5, lane = threadIdx.x & 31;
  if (lane == 0) red[wid] = ss;
  __syncthreads();
  if (threadIdx.x == 0) {
    float t = 0.f;
    for (int i = 0; i < 8; ++i) t += red[i];
    red[0] = rsqrtf(t / (float)DMODEL + 1.1920929e-07f);
  }
  __syncthreads();
  float s = red[0];
  for (int i = threadIdx.x; i < DMODEL; i += 256)
    out[(size_t)row * DMODEL + i] = (_Float16)(xr[i] * s * w[i]);
}

// ---------------- WMMA GEMM: out = A[M,K] * W[N,K]^T + bias ----------------
// 128x128 block tile, 8 waves (4m x 2n), 8 accumulators/wave, double-buffered LDS,
// global_prefetch_b8 for the k+2 panels.
#define GLDK 40  // 32 + 8 pad halves

__device__ __forceinline__ void gemm_stage(const _Float16* __restrict__ A,
                                           const _Float16* __restrict__ W,
                                           _Float16* As, _Float16* Bs,
                                           int m0, int n0, int k0, int N, int K, int tid) {
  {  // A: 128 rows x 32 halves
    int r = tid >> 1, c = (tid & 1) * 16;
    const _Float16* g = A + (size_t)(m0 + r) * K + k0 + c;
    copy8(g, As + r * GLDK + c);
    copy8(g + 8, As + r * GLDK + c + 8);
  }
  {  // B[n][k] straight from W[N,K]
    int n = tid >> 1, kp = (tid & 1) * 16;
    int gn = n0 + n;
    if (gn < N) {
      const _Float16* g = W + (size_t)gn * K + k0 + kp;
      copy8(g, Bs + n * GLDK + kp);
      copy8(g + 8, Bs + n * GLDK + kp + 8);
    } else {
      v8h zz = (v8h)((_Float16)0.f);
      *(v8h*)(Bs + n * GLDK + kp) = zz;
      *(v8h*)(Bs + n * GLDK + kp + 8) = zz;
    }
  }
}

template <int EP>
__global__ void __launch_bounds__(256) k_gemm(const _Float16* __restrict__ A,
                                              const _Float16* __restrict__ W,
                                              const float* __restrict__ bias,
                                              int M, int N, int K,
                                              float* __restrict__ outF,
                                              _Float16* __restrict__ outH1,
                                              _Float16* __restrict__ outH2,
                                              const float* __restrict__ resid,
                                              const float* __restrict__ gate) {
  constexpr int BM = 128, BN = 128, BK = 32;
  __shared__ __align__(16) _Float16 As[2][BM * GLDK];
  __shared__ __align__(16) _Float16 Bs[2][BN * GLDK];
  int m0 = blockIdx.y * BM, n0 = blockIdx.x * BN;
  int tid = threadIdx.x, lane = tid & 31, wid = tid >> 5;
  int wm = wid & 3, wn = wid >> 2;  // wave covers rows [wm*32, wm*32+32), cols [wn*64, +64)
  v8f acc[2][4];
#pragma unroll
  for (int mt = 0; mt < 2; ++mt)
#pragma unroll
    for (int i = 0; i < 4; ++i)
#pragma unroll
      for (int j = 0; j < 8; ++j) acc[mt][i][j] = 0.f;

  gemm_stage(A, W, As[0], Bs[0], m0, n0, 0, N, K, tid);
  __syncthreads();
  int buf = 0;
  for (int k0 = 0; k0 < K; k0 += BK) {
    int nk = k0 + BK;
    if (nk < K) gemm_stage(A, W, As[buf ^ 1], Bs[buf ^ 1], m0, n0, nk, N, K, tid);
    if (k0 + 2 * BK < K) {  // global_prefetch_b8 of the k+2 panels
      __builtin_prefetch(A + (size_t)(m0 + (tid >> 1)) * K + k0 + 2 * BK, 0, 0);
      int gn = n0 + (tid >> 1);
      if (gn < N) __builtin_prefetch(W + (size_t)gn * K + k0 + 2 * BK, 0, 0);
    }
    v16h af0 = frag_row(As[buf], GLDK, wm * 32, 0, lane);
    v16h af1 = frag_row(As[buf], GLDK, wm * 32 + 16, 0, lane);
#pragma unroll
    for (int nt = 0; nt < 4; ++nt) {
      v16h bf = frag_kc(Bs[buf], GLDK, 0, wn * 64 + nt * 16, lane);
      acc[0][nt] = wmma_f32(af0, bf, acc[0][nt]);
      acc[1][nt] = wmma_f32(af1, bf, acc[1][nt]);
    }
    __syncthreads();
    buf ^= 1;
  }

  int nbase = n0 + wn * 64;
#pragma unroll
  for (int mt = 0; mt < 2; ++mt) {
    int mbase = m0 + wm * 32 + mt * 16 + ((lane >> 4) << 3);
#pragma unroll
    for (int nt = 0; nt < 4; ++nt) {
      int n = nbase + nt * 16 + (lane & 15);
      if (n >= N) continue;
      float bv = bias[n];
#pragma unroll
      for (int j = 0; j < 8; ++j) {
        int m = mbase + j;
        float v = acc[mt][nt][j] + bv;
        if (EP == 0) {
          if (n < DINNER) outH1[(size_t)m * DINNER + n] = (_Float16)v;
          else            outH2[(size_t)m * DINNER + (n - DINNER)] = (_Float16)v;
        } else if (EP == 1) {
          outH1[(size_t)m * N + n] = (_Float16)v;
        } else {
          v = fminf(fmaxf(v, -100.f), 100.f);
          outF[(size_t)m * N + n] = resid[(size_t)m * N + n] + v * gate[0];
        }
      }
    }
  }
}

// ---------------- depthwise causal conv(4) + SiLU ----------------
__global__ void k_conv_silu(const _Float16* __restrict__ x, const float* __restrict__ cw,
                            const float* __restrict__ cb, _Float16* __restrict__ out) {
  size_t i = (size_t)blockIdx.x * blockDim.x + threadIdx.x;
  if (i >= (size_t)NTOK * DINNER) return;
  int ch = (int)(i % DINNER);
  size_t m = i / DINNER;
  int l = (int)(m % SEQL);
  size_t base = (m - (size_t)l) * DINNER + ch;
  float acc = cb[ch];
#pragma unroll
  for (int k = 0; k < 4; ++k) {
    int ll = l - 3 + k;
    if (ll >= 0) acc += cw[ch * 4 + k] * (float)x[base + (size_t)ll * DINNER];
  }
  float s = acc / (1.f + __expf(-acc));
  out[i] = (_Float16)s;
}

// ---------------- per-tile transpose of X: xs_h[token][ch] -> XT[tile][d][c] ----------------
__global__ void __launch_bounds__(256) k_transpose_x(const _Float16* __restrict__ Xs,
                                                     _Float16* __restrict__ XT) {
  __shared__ __align__(16) _Float16 T[64 * LDT];  // [c][d]
  int tidx = blockIdx.x;
  int n = tidx % NCHUNK, h = (tidx / NCHUNK) % NHEADS, b = tidx / (NCHUNK * NHEADS);
  int t = threadIdx.x;
  {
    int c = t >> 2, part = (t & 3) * 16;
    const _Float16* g = Xs + ((size_t)b * SEQL + (size_t)n * 64 + c) * DINNER + h * 64 + part;
    copy8(g, T + c * LDT + part);
    copy8(g + 8, T + c * LDT + part + 8);
  }
  __syncthreads();
  {
    int d = t >> 2, c0 = (t & 3) * 16;
    _Float16* g = XT + (size_t)tidx * 4096 + (size_t)d * 64 + c0;
#pragma unroll
    for (int i = 0; i < 16; ++i) g[i] = T[(c0 + i) * LDT + d];
  }
}

// ---------------- tile-based split of xp: A(t), normalized B (both layouts), C ----------------
__global__ void __launch_bounds__(256) k_split_norm(const _Float16* __restrict__ xp,
                                                    const float* __restrict__ A_log,
                                                    float* __restrict__ Aval,
                                                    _Float16* __restrict__ BnN,
                                                    _Float16* __restrict__ BnT,
                                                    _Float16* __restrict__ Cn) {
  __shared__ __align__(16) _Float16 Bl[64 * LDT];  // [c][s]
  __shared__ __align__(16) _Float16 Cl[64 * LDT];  // [c][s]
  int tidx = blockIdx.x;
  int n = tidx % NCHUNK, h = (tidx / NCHUNK) % NHEADS, b = tidx / (NCHUNK * NHEADS);
  int lane = threadIdx.x & 31, wid = threadIdx.x >> 5;
  for (int q = 0; q < 8; ++q) {
    int c = wid * 8 + q;
    size_t m = (size_t)b * SEQL + (size_t)n * 64 + c;
    const _Float16* row = xp + m * PROJ + h * (1 + 2 * DSTATE);
    {
      float v0 = (float)row[1 + 2 * lane], v1 = (float)row[2 + 2 * lane];
      float ss = v0 * v0 + v1 * v1;
      for (int mk = 16; mk > 0; mk >>= 1) ss += __shfl_xor(ss, mk, 32);
      float inv = 1.f / fmaxf(sqrtf(ss), 1e-12f);
      Bl[c * LDT + 2 * lane] = (_Float16)(v0 * inv);
      Bl[c * LDT + 2 * lane + 1] = (_Float16)(v1 * inv);
    }
    {
      float v0 = (float)row[65 + 2 * lane], v1 = (float)row[66 + 2 * lane];
      float ss = v0 * v0 + v1 * v1;
      for (int mk = 16; mk > 0; mk >>= 1) ss += __shfl_xor(ss, mk, 32);
      float inv = 1.f / fmaxf(sqrtf(ss), 1e-12f);
      Cl[c * LDT + 2 * lane] = (_Float16)(v0 * inv);
      Cl[c * LDT + 2 * lane + 1] = (_Float16)(v1 * inv);
    }
    if (lane == 0) {
      float dt = (float)row[0];
      dt = fminf(fmaxf(dt, -10.f), 10.f);
      float sp = logf(1.f + __expf(dt));
      sp = fminf(fmaxf(sp, 0.01f), 10.f);
      float a = fminf(fmaxf(A_log[h], -2.3f), -0.01f) * sp;
      Aval[m * NHEADS + h] = fminf(fmaxf(a, -20.f), -0.001f);
    }
  }
  __syncthreads();
  size_t tile = (size_t)tidx * 4096;
  {  // natural layout out (coalesced)
    int c = threadIdx.x >> 2, s0 = (threadIdx.x & 3) * 16;
    copy8(Bl + c * LDT + s0, BnN + tile + (size_t)c * 64 + s0);
    copy8(Bl + c * LDT + s0 + 8, BnN + tile + (size_t)c * 64 + s0 + 8);
    copy8(Cl + c * LDT + s0, Cn + tile + (size_t)c * 64 + s0);
    copy8(Cl + c * LDT + s0 + 8, Cn + tile + (size_t)c * 64 + s0 + 8);
  }
  {  // transposed B out
    int s = threadIdx.x >> 2, c0 = (threadIdx.x & 3) * 16;
    _Float16* g = BnT + tile + (size_t)s * 64 + c0;
#pragma unroll
    for (int i = 0; i < 16; ++i) g[i] = Bl[(c0 + i) * LDT + s];
  }
}

// stage one 64x64 f16 tile (contiguous global) into LDS with LDT stride
__device__ __forceinline__ void stage_tile_128(const _Float16* g, _Float16* s, int tid) {
  int r = tid >> 1, part = (tid & 1) * 32;
#pragma unroll
  for (int i = 0; i < 4; ++i)
    copy8(g + (size_t)r * 64 + part + i * 8, s + r * LDT + part + i * 8);
}

// ---------------- chunk surprise: mean((X^T B)^2) per (b,h,n) ----------------
__global__ void __launch_bounds__(128) k_surprise(const _Float16* __restrict__ BnT,
                                                  const _Float16* __restrict__ XT,
                                                  float* __restrict__ surp) {
  __shared__ __align__(16) _Float16 Xl[64 * LDT];  // [d][c]
  __shared__ __align__(16) _Float16 Btl[64 * LDT]; // [s][c]
  __shared__ float red[4];
  int tidx = blockIdx.x;
  int lane = threadIdx.x & 31, wid = threadIdx.x >> 5;
  stage_tile_128(XT + (size_t)tidx * 4096, Xl, threadIdx.x);
  stage_tile_128(BnT + (size_t)tidx * 4096, Btl, threadIdx.x);
  __syncthreads();
  float ssum = 0.f;
#pragma unroll
  for (int st = 0; st < 4; ++st) {  // P[d,s] = sum_c X[c,d] B[c,s]
    v8f acc;
#pragma unroll
    for (int j = 0; j < 8; ++j) acc[j] = 0.f;
#pragma unroll
    for (int k0 = 0; k0 < 64; k0 += 32) {
      v16h af = frag_row(Xl, LDT, wid * 16, k0, lane);
      v16h bf = frag_kc(Btl, LDT, k0, st * 16, lane);
      acc = wmma_f32(af, bf, acc);
    }
#pragma unroll
    for (int j = 0; j < 8; ++j) ssum += acc[j] * acc[j];
  }
  for (int m = 16; m > 0; m >>= 1) ssum += __shfl_xor(ssum, m, 32);
  if (lane == 0) red[wid] = ssum;
  __syncthreads();
  if (threadIdx.x == 0) surp[tidx] = (red[0] + red[1] + red[2] + red[3]) * (1.f / 4096.f);
}

// ---------------- per-head EMA + alpha ----------------
__global__ void __launch_bounds__(256) k_alpha(const float* __restrict__ surp,
                                               const float* __restrict__ ema0,
                                               const float* __restrict__ l2ab,
                                               const float* __restrict__ l2b,
                                               float* __restrict__ alpha) {
  __shared__ float red[8];
  __shared__ float emaS;
  int h = blockIdx.x, t = threadIdx.x;
  int b = t >> 7, n = t & 127;
  size_t idx = (((size_t)b * NHEADS) + h) * NCHUNK + n;
  float s = surp[idx];
  float v = s;
  for (int m = 16; m > 0; m >>= 1) v += __shfl_xor(v, m, 32);
  if ((t & 31) == 0) red[t >> 5] = v;
  __syncthreads();
  if (t == 0) {
    float tot = 0.f;
    for (int i = 0; i < 8; ++i) tot += red[i];
    emaS = 0.99f * ema0[h] + 0.01f * (tot / 256.f);
  }
  __syncthreads();
  float norm = s / (emaS + 1e-6f);
  float ab = 1.f - exp2f(fminf(fmaxf(l2ab[h], -3.32f), -0.015f));
  float beta = exp2f(fminf(fmaxf(l2b[h], -2.f), 2.f));
  float boost = fmaxf(tanhf(beta * norm), 0.f);
  alpha[idx] = fminf(fmaxf(ab + (1.f - ab) * boost, 0.01f), 0.999f);
}

// ---------------- A_mod cumsum per tile ----------------
__global__ void k_cumsum(const float* __restrict__ Aval, const float* __restrict__ alpha,
                         float* __restrict__ acs, float* __restrict__ dchunk) {
  int tidx = blockIdx.x * 64 + threadIdx.x;
  if (tidx >= NTILES) return;
  int n = tidx % NCHUNK, h = (tidx / NCHUNK) % NHEADS, b = tidx / (NCHUNK * NHEADS);
  float oma = 1.f - alpha[tidx];
  size_t base = ((size_t)b * SEQL + (size_t)n * 64) * NHEADS + h;
  float cs = 0.f;
  for (int c = 0; c < 64; ++c) {
    cs += Aval[base + (size_t)c * NHEADS] * oma;
    acs[(size_t)tidx * 64 + c] = cs;
  }
  dchunk[tidx] = __expf(cs);
}

// ---------------- inter-chunk transposed state scan ----------------
// state'[d][s]; hcf'[d,s] = sum_c X[c,d] * (dte[c]*B[c,s]) -> A=XT, B=scaled BnT
__global__ void __launch_bounds__(128) k_scan(const _Float16* __restrict__ BnT,
                                              const _Float16* __restrict__ XT,
                                              const float* __restrict__ acs,
                                              const float* __restrict__ dchunk,
                                              _Float16* __restrict__ HT) {
  __shared__ __align__(16) _Float16 Xl[64 * LDT];   // [d][c]
  __shared__ __align__(16) _Float16 Bwl[64 * LDT];  // [s][c] * dte[c]
  __shared__ __align__(16) _Float16 dteh[64];
  int bh = blockIdx.x;
  int lane = threadIdx.x & 31, wid = threadIdx.x >> 5;
  v8f carry[4];
#pragma unroll
  for (int i = 0; i < 4; ++i)
#pragma unroll
    for (int j = 0; j < 8; ++j) carry[i][j] = 0.f;

  for (int n = 0; n < NCHUNK; ++n) {
    size_t tidx = (size_t)bh * NCHUNK + n;
    if (n + 1 < NCHUNK) {  // global_prefetch_b8: next chunk's tiles while this one computes
      __builtin_prefetch(BnT + (tidx + 1) * 4096 + (size_t)threadIdx.x * 32, 0, 0);
      __builtin_prefetch(XT + (tidx + 1) * 4096 + (size_t)threadIdx.x * 32, 0, 0);
    }
    {  // HT[n] = carry (pre-update); element (d=row, s=col)
      _Float16* H = HT + tidx * 4096;
      int drow = wid * 16 + ((lane >> 4) << 3);
#pragma unroll
      for (int st = 0; st < 4; ++st) {
        int s = st * 16 + (lane & 15);
#pragma unroll
        for (int j = 0; j < 8; ++j) H[(size_t)(drow + j) * 64 + s] = (_Float16)carry[st][j];
      }
    }
    if (threadIdx.x < 64) {
      float last = acs[tidx * 64 + 63];
      dteh[threadIdx.x] = (_Float16)__expf(last - acs[tidx * 64 + threadIdx.x]);
    }
    __syncthreads();
    {  // stage X and decay-weighted B (v_pk_mul_f16 on v8h)
      int r = threadIdx.x >> 1, part = (threadIdx.x & 1) * 32;
      const _Float16* xg = XT + tidx * 4096 + (size_t)r * 64 + part;
      const _Float16* bg = BnT + tidx * 4096 + (size_t)r * 64 + part;
#pragma unroll
      for (int i = 0; i < 4; ++i) {
        copy8(xg + i * 8, Xl + r * LDT + part + i * 8);
        v8h bv = *(const v8h*)(bg + i * 8);
        v8h wv = *(const v8h*)(dteh + part + i * 8);
        *(v8h*)(Bwl + r * LDT + part + i * 8) = bv * wv;
      }
    }
    __syncthreads();
    float dch = dchunk[tidx];
#pragma unroll
    for (int st = 0; st < 4; ++st) {
      v8f acc;
#pragma unroll
      for (int j = 0; j < 8; ++j) acc[j] = 0.f;
#pragma unroll
      for (int k0 = 0; k0 < 64; k0 += 32) {
        v16h af = frag_row(Xl, LDT, wid * 16, k0, lane);
        v16h bf = frag_kc(Bwl, LDT, k0, st * 16, lane);
        acc = wmma_f32(af, bf, acc);
      }
#pragma unroll
      for (int j = 0; j < 8; ++j) carry[st][j] = dch * carry[st][j] + acc[j];
    }
    __syncthreads();
  }
}

// ---------------- fused intra+inter output per tile ----------------
__global__ void __launch_bounds__(128) k_y(const _Float16* __restrict__ BnN,
                                           const _Float16* __restrict__ Cn,
                                           const _Float16* __restrict__ XT,
                                           const _Float16* __restrict__ HT,
                                           const float* __restrict__ acs,
                                           const _Float16* __restrict__ z,
                                           _Float16* __restrict__ yout) {
  __shared__ __align__(16) _Float16 Cl[64 * LDT];   // [i][s]
  __shared__ __align__(16) _Float16 Bl[64 * LDT];   // [j][s]
  __shared__ __align__(16) _Float16 Xl[64 * LDT];   // [d][j]
  __shared__ __align__(16) _Float16 Hl[64 * LDT];   // [d][s]
  __shared__ __align__(16) _Float16 Wl[64 * LDT];   // [i][j]
  __shared__ float acsS[64];
  int tidx = blockIdx.x;
  int n = tidx % NCHUNK, h = (tidx / NCHUNK) % NHEADS, b = tidx / (NCHUNK * NHEADS);
  int lane = threadIdx.x & 31, wid = threadIdx.x >> 5;
  stage_tile_128(Cn + (size_t)tidx * 4096, Cl, threadIdx.x);
  stage_tile_128(BnN + (size_t)tidx * 4096, Bl, threadIdx.x);
  stage_tile_128(XT + (size_t)tidx * 4096, Xl, threadIdx.x);
  stage_tile_128(HT + (size_t)tidx * 4096, Hl, threadIdx.x);
  if (threadIdx.x < 64) acsS[threadIdx.x] = acs[(size_t)tidx * 64 + threadIdx.x];
  __syncthreads();
  // Wl[i][j] = Lmat[i,j] * (C B^T)[i,j] ; CB: A=C[i][s], B=Bn[j][s] ([n][k] layout)
#pragma unroll
  for (int jt = 0; jt < 4; ++jt) {
    v8f acc;
#pragma unroll
    for (int j = 0; j < 8; ++j) acc[j] = 0.f;
#pragma unroll
    for (int k0 = 0; k0 < 64; k0 += 32) {
      v16h af = frag_row(Cl, LDT, wid * 16, k0, lane);
      v16h bf = frag_kc(Bl, LDT, k0, jt * 16, lane);
      acc = wmma_f32(af, bf, acc);
    }
    int i0 = wid * 16 + ((lane >> 4) << 3);
    int jc = jt * 16 + (lane & 15);
#pragma unroll
    for (int j = 0; j < 8; ++j) {
      int i = i0 + j;
      float lm = (i >= jc) ? __expf(acsS[i] - acsS[jc]) : 0.f;
      Wl[i * LDT + jc] = (_Float16)(acc[j] * lm);
    }
  }
  __syncthreads();
  size_t zbase = ((size_t)b * SEQL + (size_t)n * 64) * DINNER + h * 64;
#pragma unroll
  for (int dt_ = 0; dt_ < 4; ++dt_) {
    v8f acc;
#pragma unroll
    for (int j = 0; j < 8; ++j) acc[j] = 0.f;
#pragma unroll
    for (int k0 = 0; k0 < 64; k0 += 32) {  // Y_inter = C * h_inter : B operand Hl[d][s]
      v16h af = frag_row(Cl, LDT, wid * 16, k0, lane);
      v16h bf = frag_kc(Hl, LDT, k0, dt_ * 16, lane);
      acc = wmma_f32(af, bf, acc);
    }
    int i0 = wid * 16 + ((lane >> 4) << 3);
#pragma unroll
    for (int j = 0; j < 8; ++j) acc[j] *= __expf(acsS[i0 + j]);  // decay_from_start
#pragma unroll
    for (int k0 = 0; k0 < 64; k0 += 32) {  // += Y_intra = W * X : B operand Xl[d][j]
      v16h af = frag_row(Wl, LDT, wid * 16, k0, lane);
      v16h bf = frag_kc(Xl, LDT, k0, dt_ * 16, lane);
      acc = wmma_f32(af, bf, acc);
    }
    int dd = dt_ * 16 + (lane & 15);
#pragma unroll
    for (int j = 0; j < 8; ++j) {
      int i = i0 + j;
      float yv = fminf(fmaxf(acc[j], -100.f), 100.f);
      float zv = (float)z[zbase + (size_t)i * DINNER + dd];
      float sg = 1.f / (1.f + __expf(-zv));
      yout[zbase + (size_t)i * DINNER + dd] = (_Float16)(yv * sg);
    }
  }
}

// ---------------- launcher ----------------
extern "C" void kernel_launch(void* const* d_in, const int* in_sizes, int n_in,
                              void* d_out, int out_size, void* d_ws, size_t ws_size,
                              hipStream_t stream) {
  (void)in_sizes; (void)n_in; (void)out_size; (void)ws_size;
  char* ws = (char*)d_ws;
  size_t off = 0;
  auto alloc = [&](size_t elems, size_t esz) -> void* {
    void* p = ws + off;
    off += ((elems * esz + 255) / 256) * 256;
    return p;
  };
  _Float16* w_in_h  = (_Float16*)alloc((size_t)2 * DINNER * DMODEL, 2);
  _Float16* w_xp_h  = (_Float16*)alloc((size_t)PROJ * DINNER, 2);
  _Float16* w_out_h = (_Float16*)alloc((size_t)DMODEL * DINNER, 2);
  _Float16* xn_h    = (_Float16*)alloc((size_t)NTOK * DMODEL, 2);
  _Float16* xpre_h  = (_Float16*)alloc((size_t)NTOK * DINNER, 2);
  _Float16* z_h     = (_Float16*)alloc((size_t)NTOK * DINNER, 2);
  _Float16* xs_h    = (_Float16*)alloc((size_t)NTOK * DINNER, 2);
  _Float16* xp_h    = (_Float16*)alloc((size_t)NTOK * PROJ, 2);
  _Float16* XT_h    = (_Float16*)alloc((size_t)NTILES * 4096, 2);
  _Float16* BnN_h   = (_Float16*)alloc((size_t)NTILES * 4096, 2);
  _Float16* BnT_h   = (_Float16*)alloc((size_t)NTILES * 4096, 2);
  _Float16* Cn_h    = (_Float16*)alloc((size_t)NTILES * 4096, 2);
  _Float16* HT_h    = (_Float16*)alloc((size_t)NTILES * 4096, 2);
  _Float16* y_h     = (_Float16*)alloc((size_t)NTOK * DINNER, 2);
  float*    Aval    = (float*)alloc((size_t)NTOK * NHEADS, 4);
  float*    surp    = (float*)alloc(NTILES, 4);
  float*    alphaF  = (float*)alloc(NTILES, 4);
  float*    acsF    = (float*)alloc((size_t)NTILES * 64, 4);
  float*    dchunk  = (float*)alloc(NTILES, 4);

  const float* hidden = (const float*)d_in[0];
  const float* norm_w = (const float*)d_in[1];
  const float* inW    = (const float*)d_in[2];
  const float* inB    = (const float*)d_in[3];
  const float* convW  = (const float*)d_in[4];
  const float* convB  = (const float*)d_in[5];
  const float* xpW    = (const float*)d_in[6];
  const float* xpB    = (const float*)d_in[7];
  const float* A_log  = (const float*)d_in[8];
  const float* l2ab   = (const float*)d_in[9];
  const float* l2b    = (const float*)d_in[10];
  const float* ema0   = (const float*)d_in[11];
  const float* outW   = (const float*)d_in[12];
  const float* outB   = (const float*)d_in[13];
  const float* gate   = (const float*)d_in[14];

  { int n = 2 * DINNER * DMODEL; k_f32_to_f16<<<(n + 255) / 256, 256, 0, stream>>>(inW, w_in_h, n); }
  { int n = PROJ * DINNER;       k_f32_to_f16<<<(n + 255) / 256, 256, 0, stream>>>(xpW, w_xp_h, n); }
  { int n = DMODEL * DINNER;     k_f32_to_f16<<<(n + 255) / 256, 256, 0, stream>>>(outW, w_out_h, n); }

  k_rmsnorm<<<NTOK, 256, 0, stream>>>(hidden, norm_w, xn_h);

  {
    dim3 g(2 * DINNER / 128, NTOK / 128);
    k_gemm<0><<<g, 256, 0, stream>>>(xn_h, w_in_h, inB, NTOK, 2 * DINNER, DMODEL,
                                     nullptr, xpre_h, z_h, nullptr, nullptr);
  }
  {
    size_t tot = (size_t)NTOK * DINNER;
    k_conv_silu<<<(unsigned)((tot + 255) / 256), 256, 0, stream>>>(xpre_h, convW, convB, xs_h);
  }
  k_transpose_x<<<NTILES, 256, 0, stream>>>(xs_h, XT_h);
  {
    dim3 g((PROJ + 127) / 128, NTOK / 128);
    k_gemm<1><<<g, 256, 0, stream>>>(xs_h, w_xp_h, xpB, NTOK, PROJ, DINNER,
                                     nullptr, xp_h, nullptr, nullptr, nullptr);
  }
  k_split_norm<<<NTILES, 256, 0, stream>>>(xp_h, A_log, Aval, BnN_h, BnT_h, Cn_h);
  k_surprise<<<NTILES, 128, 0, stream>>>(BnT_h, XT_h, surp);
  k_alpha<<<NHEADS, 256, 0, stream>>>(surp, ema0, l2ab, l2b, alphaF);
  k_cumsum<<<NTILES / 64, 64, 0, stream>>>(Aval, alphaF, acsF, dchunk);
  k_scan<<<BSZ * NHEADS, 128, 0, stream>>>(BnT_h, XT_h, acsF, dchunk, HT_h);
  k_y<<<NTILES, 128, 0, stream>>>(BnN_h, Cn_h, XT_h, HT_h, acsF, z_h, y_h);
  {
    dim3 g(DMODEL / 128, NTOK / 128);
    k_gemm<2><<<g, 256, 0, stream>>>(y_h, w_out_h, outB, NTOK, DMODEL, DINNER,
                                     (float*)d_out, nullptr, nullptr, hidden, gate);
  }
}
